// Attention_66314295050336
// MI455X (gfx1250) — matrix operand order repaired and checked by reading, not compiled, not run on hardware
//
#include <hip/hip_runtime.h>
#include <hip/hip_bf16.h>

// MI455X / gfx1250, wave32. All matmuls via v_wmma_f32_16x16x32_f16.
// fp32 inputs are converted once to f16 in workspace so GEMM inner loops are
// pure b128 loads -> wmma (no per-load cvt/wait serialization).

typedef __attribute__((ext_vector_type(16))) _Float16 v16h;
typedef __attribute__((ext_vector_type(8)))  _Float16 v8h;
typedef __attribute__((ext_vector_type(4)))  _Float16 v4h;
typedef __attribute__((ext_vector_type(8)))  float    v8f;

static __device__ __forceinline__ v8f wmma_f16(v16h a, v16h b, v8f c) {
  return __builtin_amdgcn_wmma_f32_16x16x32_f16(false, a, false, b, (short)0, c, false, false);
}

// ---- fragment loaders (layouts per CDNA5 ISA 7.12.2) ----

// A 16x32 f16 frag: lane row = m0+lane%16; elems 0..7 -> K = k0+hi*8+i; 8..15 -> +16.
// p0 points at (row, k0 + hi*8); both halves are 16B-aligned b128 loads.
static __device__ __forceinline__ v16h load_a_f16(const _Float16* p0) {
  v8h lo  = *(const v8h*)(p0);
  v8h hi8 = *(const v8h*)(p0 + 16);
  v16h a;
#pragma unroll
  for (int i = 0; i < 8; ++i) { a[i] = lo[i]; a[8 + i] = hi8[i]; }
  return a;
}

// B 32x16 f16 frag: column n = lane%16; elems 0..7 -> K = hi*8+i; 8..15 -> +16.
// p points at (col, k0 + hi*16); loads from p - hi*8 and +16 (16B-aligned b128).
static __device__ __forceinline__ v16h load_b_f16(const _Float16* p) {
  p -= ((threadIdx.x & 31) >> 4) * 8;
  v8h lo  = *(const v8h*)(p);
  v8h hi8 = *(const v8h*)(p + 16);
  v16h b;
#pragma unroll
  for (int i = 0; i < 8; ++i) { b[i] = lo[i]; b[8 + i] = hi8[i]; }
  return b;
}

// =====================  Kernel 0: one-shot f32 -> f16 convert  =====================
__global__ __launch_bounds__(256) void cvt_f32_f16(const float* __restrict__ in,
                                                   _Float16* __restrict__ out, int n) {
  const int i = (blockIdx.x * 256 + threadIdx.x) * 4;
  if (i + 3 < n) {
    const float4 v = *(const float4*)(in + i);
    v4h h;
    h[0] = (_Float16)v.x; h[1] = (_Float16)v.y;
    h[2] = (_Float16)v.z; h[3] = (_Float16)v.w;
    *(v4h*)(out + i) = h;
  }
}

// =====================  Kernel 1: QKV projection  =====================
// xh[8192,384] f16, wh[1152,384] f16 -> Q,K f16 [B*H,1024,64], Vt f16 [B*H,64,1024]
__global__ __launch_bounds__(256) void qkv_kernel(const _Float16* __restrict__ xh,
                                                  const _Float16* __restrict__ wh,
                                                  const float* __restrict__ bqkv,
                                                  _Float16* __restrict__ Q,
                                                  _Float16* __restrict__ K,
                                                  _Float16* __restrict__ Vt) {
  const int Kd = 384;
  const int lane = threadIdx.x & 31, ln = lane & 15, hi = lane >> 4;
  const int wave = blockIdx.x * 8 + (threadIdx.x >> 5);    // 9216 waves
  const int tm = wave / 18, tn = wave - tm * 18;           // 512 x 18 tiles
  const int m0 = tm * 16, n0 = tn * 64;

  const _Float16* ap = xh + (size_t)(m0 + ln) * Kd + hi * 8;
  const _Float16* bp = wh + (size_t)(n0 + ln) * Kd + hi * 16;

  v8f c0 = {}, c1 = {}, c2 = {}, c3 = {};
  for (int k0 = 0; k0 < Kd; k0 += 32) {
    v16h a  = load_a_f16(ap + k0);
    v16h b0 = load_b_f16(bp + k0);
    v16h b1 = load_b_f16(bp + 16 * Kd + k0);
    v16h b2 = load_b_f16(bp + 32 * Kd + k0);
    v16h b3 = load_b_f16(bp + 48 * Kd + k0);
    c0 = wmma_f16(a, b0, c0);
    c1 = wmma_f16(a, b1, c1);
    c2 = wmma_f16(a, b2, c2);
    c3 = wmma_f16(a, b3, c3);
  }

  v8f cs[4] = {c0, c1, c2, c3};
#pragma unroll
  for (int j = 0; j < 4; ++j) {
    const int o = n0 + j * 16 + ln;            // output channel in [0,1152)
    const int which = o / 384;                 // 0=Q 1=K 2=V (uniform per tile)
    const int cc = o - which * 384;
    const int h = cc >> 6, d = cc & 63;
    const float bias = bqkv[o];
#pragma unroll
    for (int r = 0; r < 8; ++r) {
      const int m = m0 + r + 8 * hi;           // token index in [0,8192)
      const int bb = m >> 10, nn = m & 1023;
      const int head = bb * 6 + h;
      const _Float16 hv = (_Float16)(cs[j][r] + bias);
      if (which == 0)      Q [((size_t)head * 1024 + nn) * 64 + d] = hv;
      else if (which == 1) K [((size_t)head * 1024 + nn) * 64 + d] = hv;
      else                 Vt[((size_t)head * 64 + d) * 1024 + nn] = hv;
    }
  }
}

// =====================  Kernel 2: fused flash attention  =====================
// Per wave: 16 query rows x D=64 output for one head; online softmax with
// threshold pruning (score*scale < 0 -> -10000) over all 1024 keys.
__global__ __launch_bounds__(256) void attn_kernel(const _Float16* __restrict__ Q,
                                                   const _Float16* __restrict__ Km,
                                                   const _Float16* __restrict__ Vt,
                                                   _Float16* __restrict__ O) {
  const float scale = 0.125f;                       // 64^-0.5
  const int lane = threadIdx.x & 31, ln = lane & 15, hi = lane >> 4;
  const int wib = threadIdx.x >> 5;
  const int wave = blockIdx.x * 8 + wib;            // 3072 waves
  const int head = wave >> 6;                       // 48 heads
  const int m0 = (wave & 63) << 4;
  const int bb = head / 6, h = head - bb * 6;

  __shared__ __align__(16) _Float16 pbuf[8][16 * 32]; // 1KB slab per wave
  _Float16* myp = pbuf[wib];

  const _Float16* qbase = Q  + (size_t)head * 1024 * 64;
  const _Float16* kbase = Km + (size_t)head * 1024 * 64;
  const _Float16* vbase = Vt + (size_t)head * 64 * 1024;

  // Q A-fragments (reused across whole key loop)
  const v16h aq0 = load_a_f16(qbase + (m0 + ln) * 64 + hi * 8);
  const v16h aq1 = load_a_f16(qbase + (m0 + ln) * 64 + 32 + hi * 8);

  v8f o0 = {}, o1 = {}, o2 = {}, o3 = {};
  float mrow[8], lrow[8];
#pragma unroll
  for (int r = 0; r < 8; ++r) { mrow[r] = -1e30f; lrow[r] = 0.0f; }

  for (int jb = 0; jb < 1024; jb += 32) {
    // --- S tile (16 rows x 32 keys): B-frags are K^T (feature-major contiguous)
    v16h bk00 = load_b_f16(kbase + (jb      + ln) * 64 +      hi * 16);
    v16h bk01 = load_b_f16(kbase + (jb      + ln) * 64 + 32 + hi * 16);
    v16h bk10 = load_b_f16(kbase + (jb + 16 + ln) * 64 +      hi * 16);
    v16h bk11 = load_b_f16(kbase + (jb + 16 + ln) * 64 + 32 + hi * 16);
    v8f s0 = {}, s1 = {};
    s0 = wmma_f16(aq0, bk00, s0); s0 = wmma_f16(aq1, bk01, s0);
    s1 = wmma_f16(aq0, bk10, s1); s1 = wmma_f16(aq1, bk11, s1);

    if (jb + 32 < 1024) {     // prefetch next K/V tile (gfx1250 global_prefetch_b8)
      __builtin_prefetch(kbase + (jb + 32 + ln) * 64, 0, 0);
      __builtin_prefetch(vbase + ln * 1024 + jb + 32, 0, 0);
    }

    // --- scale, threshold-prune, online softmax in C-fragment layout
    float corr[8];
#pragma unroll
    for (int r = 0; r < 8; ++r) {
      float v0 = s0[r] * scale; v0 = (v0 < 0.0f) ? -10000.0f : v0;
      float v1 = s1[r] * scale; v1 = (v1 < 0.0f) ? -10000.0f : v1;
      float t = fmaxf(v0, v1);                  // row max over 32 keys:
      t = fmaxf(t, __shfl_xor(t, 1));           // butterfly within 16-lane half
      t = fmaxf(t, __shfl_xor(t, 2));
      t = fmaxf(t, __shfl_xor(t, 4));
      t = fmaxf(t, __shfl_xor(t, 8));
      const float mnew = fmaxf(mrow[r], t);
      const float c = __expf(mrow[r] - mnew);
      mrow[r] = mnew; corr[r] = c;
      const float p0 = __expf(v0 - mnew);
      const float p1 = __expf(v1 - mnew);
      float ps = p0 + p1;
      ps += __shfl_xor(ps, 1);
      ps += __shfl_xor(ps, 2);
      ps += __shfl_xor(ps, 4);
      ps += __shfl_xor(ps, 8);
      lrow[r] = lrow[r] * c + ps;
      // stage P (C layout -> LDS, row-major 16x32 f16)
      myp[(r + 8 * hi) * 32 +      ln] = (_Float16)p0;
      myp[(r + 8 * hi) * 32 + 16 + ln] = (_Float16)p1;
      o0[r] *= c; o1[r] *= c; o2[r] *= c; o3[r] *= c;
    }

    // --- reload P as A-fragment (ds_load_b128 x2), accumulate O += P @ V
    v16h pa  = load_a_f16(myp + ln * 32 + hi * 8);
    v16h bv0 = load_b_f16(vbase + ( 0 + ln) * 1024 + jb + hi * 16);
    v16h bv1 = load_b_f16(vbase + (16 + ln) * 1024 + jb + hi * 16);
    v16h bv2 = load_b_f16(vbase + (32 + ln) * 1024 + jb + hi * 16);
    v16h bv3 = load_b_f16(vbase + (48 + ln) * 1024 + jb + hi * 16);
    o0 = wmma_f16(pa, bv0, o0);
    o1 = wmma_f16(pa, bv1, o1);
    o2 = wmma_f16(pa, bv2, o2);
    o3 = wmma_f16(pa, bv3, o3);
  }

  // --- finalize: O /= l, store token-major f16 [8192, 384] (c = h*64 + d)
#pragma unroll
  for (int r = 0; r < 8; ++r) {
    const float inv = 1.0f / lrow[r];
    const int row = m0 + r + 8 * hi;
    _Float16* po = O + ((size_t)(bb * 1024 + row) * 6 + h) * 64;
    po[ 0 + ln] = (_Float16)(o0[r] * inv);
    po[16 + ln] = (_Float16)(o1[r] * inv);
    po[32 + ln] = (_Float16)(o2[r] * inv);
    po[48 + ln] = (_Float16)(o3[r] * inv);
  }
}

// =====================  Kernel 3: output projection  =====================
// O f16 [8192,384] @ w_proj^T + b_proj -> out fp32 [8192,384]
__global__ __launch_bounds__(256) void proj_kernel(const _Float16* __restrict__ Oi,
                                                   const _Float16* __restrict__ wh,
                                                   const float* __restrict__ bias,
                                                   float* __restrict__ out) {
  const int Kd = 384;
  const int lane = threadIdx.x & 31, ln = lane & 15, hi = lane >> 4;
  const int wave = blockIdx.x * 8 + (threadIdx.x >> 5);    // 3072 waves
  const int tm = wave / 6, tn = wave - tm * 6;             // 512 x 6 tiles
  const int m0 = tm * 16, n0 = tn * 64;

  const _Float16* ap = Oi + (size_t)(m0 + ln) * Kd + hi * 8;
  const _Float16* bp = wh + (size_t)(n0 + ln) * Kd + hi * 16;

  v8f c0 = {}, c1 = {}, c2 = {}, c3 = {};
  for (int k0 = 0; k0 < Kd; k0 += 32) {
    v16h a  = load_a_f16(ap + k0);
    v16h b0 = load_b_f16(bp + k0);
    v16h b1 = load_b_f16(bp + 16 * Kd + k0);
    v16h b2 = load_b_f16(bp + 32 * Kd + k0);
    v16h b3 = load_b_f16(bp + 48 * Kd + k0);
    c0 = wmma_f16(a, b0, c0);
    c1 = wmma_f16(a, b1, c1);
    c2 = wmma_f16(a, b2, c2);
    c3 = wmma_f16(a, b3, c3);
  }

  const float bv0 = bias[n0 +  0 + ln];
  const float bv1 = bias[n0 + 16 + ln];
  const float bv2 = bias[n0 + 32 + ln];
  const float bv3 = bias[n0 + 48 + ln];
#pragma unroll
  for (int r = 0; r < 8; ++r) {
    const int m = m0 + r + 8 * hi;
    float* po = out + (size_t)m * 384 + n0;
    po[ 0 + ln] = c0[r] + bv0;
    po[16 + ln] = c1[r] + bv1;
    po[32 + ln] = c2[r] + bv2;
    po[48 + ln] = c3[r] + bv3;
  }
}

// =====================  launch  =====================
extern "C" void kernel_launch(void* const* d_in, const int* in_sizes, int n_in,
                              void* d_out, int out_size, void* d_ws, size_t ws_size,
                              hipStream_t stream) {
  (void)in_sizes; (void)n_in; (void)out_size;
  const float* x      = (const float*)d_in[0];   // [8,1024,384]
  const float* w_qkv  = (const float*)d_in[1];   // [1152,384]
  const float* b_qkv  = (const float*)d_in[2];   // [1152]
  const float* w_proj = (const float*)d_in[3];   // [384,384]
  const float* b_proj = (const float*)d_in[4];   // [384]
  float* out = (float*)d_out;                    // [8,1024,384] fp32

  const size_t per   = (size_t)8 * 6 * 1024 * 64;   // 3,145,728 f16 elements
  const size_t n_x   = (size_t)8192 * 384;          // 3,145,728
  const size_t n_wq  = (size_t)1152 * 384;          //   442,368
  const size_t n_wp  = (size_t)384 * 384;           //   147,456
  const size_t total = 4 * per + n_x + n_wq + n_wp; // 16,318,464 f16 = 32.6 MB
  if (ws_size < total * sizeof(_Float16)) return;

  _Float16* Q   = (_Float16*)d_ws;               // [B*H,1024,64]
  _Float16* K   = Q   + per;                     // [B*H,1024,64]
  _Float16* Vt  = K   + per;                     // [B*H,64,1024] (transposed)
  _Float16* O   = Vt  + per;                     // [8192,384] token-major
  _Float16* xh  = O   + per;                     // [8192,384]
  _Float16* wqh = xh  + n_x;                     // [1152,384]
  _Float16* wph = wqh + n_wq;                    // [384,384]

  // one-shot f32 -> f16 conversions (sizes all divisible by 1024)
  cvt_f32_f16<<<(int)(n_x  / 1024), 256, 0, stream>>>(x,      xh,  (int)n_x);
  cvt_f32_f16<<<(int)(n_wq / 1024), 256, 0, stream>>>(w_qkv,  wqh, (int)n_wq);
  cvt_f32_f16<<<(int)(n_wp / 1024), 256, 0, stream>>>(w_proj, wph, (int)n_wp);

  // 9216 wave-tiles / 8 waves per block
  qkv_kernel<<<1152, 256, 0, stream>>>(xh, wqh, b_qkv, Q, K, Vt);
  // 3072 wave-tiles
  attn_kernel<<<384, 256, 0, stream>>>(Q, K, Vt, O);
  // 3072 wave-tiles
  proj_kernel<<<384, 256, 0, stream>>>(O, wph, b_proj, out);
}
